// LangModelWithLSTM_54022098649438
// MI455X (gfx1250) — matrix operand order repaired and checked by reading, not compiled, not run on hardware
//
#include <hip/hip_runtime.h>

// ---------------------------------------------------------------------------
// CDNA5 / gfx1250 implementation of a 2-layer BiLSTM LM head.
// All large GEMMs use v_wmma_f32_16x16x32_bf16 (bf16 in, f32 accumulate).
// Weights arrive f32 and are converted once per call to bf16 in workspace.
// Every weight is [N,K] row-major, which suits the contiguous-per-lane k order
// this code uses for the WMMA B-fragment (lane n reads 16 consecutive bf16).
// ---------------------------------------------------------------------------

typedef __bf16 bf16_t;
typedef __attribute__((ext_vector_type(16))) __bf16        v16bf;
typedef __attribute__((ext_vector_type(8)))  float         v8f;
typedef __attribute__((ext_vector_type(4)))  unsigned int  u32x4;   // plain builtin
// explicit global-address-space load type: guarantees global_load_b128
// (not flat_load_b128, which falsely ties LOADcnt to DScnt on CDNA5).
// Must be a builtin vector, not HIP's class-type uint4, so assignment from an
// addrspace(1) lvalue is a primitive load (no operator= overload resolution).
typedef const __attribute__((address_space(1))) u32x4 gu32x4;

static constexpr int  kB   = 16;
static constexpr int  kS   = 512;
static constexpr int  kBS  = kB * kS;        // 8192 rows
static constexpr int  kH   = 512;
static constexpr int  kG   = 4 * kH;         // 2048 gate width
static constexpr int  kDIN = 1024;           // LM+EMB = 2H, same for both layers
static constexpr int  kHID = 1024;
static constexpr int  kC   = 5;

// -------------------------- fragment loaders --------------------------------
// A fragment (16x32 bf16, M rows striped in lane%16):
//   lane<16 : K 0..15 contiguous
//   lane>=16: K 16..31 contiguous
__device__ __forceinline__ v16bf load_a_frag(const bf16_t* base, int stride_halves, int lane) {
    int r  = lane & 15;
    int c0 = (lane < 16) ? 0 : 16;
    const bf16_t* p = base + r * stride_halves + c0;
    union { u32x4 u[2]; v16bf v; } t;
    t.u[0] = *(const u32x4*)(p);
    t.u[1] = *(const u32x4*)(p + 8);
    return t.v;
}

// B fragment (32x16 bf16) from an [N,K] row-major matrix: lane n = column,
//   lane<16 : K 0..15 contiguous ; lane>=16 : K 16..31 contiguous.
__device__ __forceinline__ v16bf load_b_frag_nk(const bf16_t* base, int stride_halves, int lane) {
    int n  = lane & 15;
    int kc = (lane < 16) ? 0 : 16;
    const bf16_t* p = base + n * stride_halves + kc;
    union { u32x4 u[2]; v16bf v; } t;
    t.u[0] = *(const u32x4*)(p);
    t.u[1] = *(const u32x4*)(p + 8);
    return t.v;
}

__device__ __forceinline__ float sigmoidf_(float x) {
    return 1.0f / (1.0f + __expf(-x));
}

// -------------------------- f32 -> bf16 convert -----------------------------
__global__ __launch_bounds__(256) void cvt_f32_bf16_kernel(const float* __restrict__ src,
                                                           bf16_t* __restrict__ dst, int n) {
    int i = blockIdx.x * 256 + threadIdx.x;
    if (i < n) dst[i] = (bf16_t)src[i];
}

// -------------------------- embed + concat ----------------------------------
// X[row, col] : col<768 -> lang_emb[row, col] ; else emb_table[x[row], col-768]
__global__ __launch_bounds__(256) void embed_concat_kernel(const int* __restrict__ x,
                                                           const float* __restrict__ lang,
                                                           const float* __restrict__ etab,
                                                           bf16_t* __restrict__ X) {
    long long i = (long long)blockIdx.x * 256 + threadIdx.x;   // over 8192*1024
    if (i >= (long long)kBS * kDIN) return;
    int col = (int)(i & (kDIN - 1));
    int row = (int)(i >> 10);
    float v;
    if (col < 768) {
        v = lang[(long long)row * 768 + col];
    } else {
        int tok = x[row];
        v = etab[(long long)tok * 256 + (col - 768)];
    }
    X[i] = (bf16_t)v;
}

// -------------------------- tiled WMMA GEMM ---------------------------------
// C[M,N] = act( A[M,K](bf16) * W[N,K]^T(bf16) + bias0 + bias1 ), C bf16.
// Block tile 128x128x32, 8 waves (4 in M, 2 in N), 32x64 per wave.
template <int ACT>
__global__ __launch_bounds__(256) void gemm_bf16_kernel(const bf16_t* __restrict__ A,
                                                        const bf16_t* __restrict__ W,
                                                        const float* __restrict__ bias0,
                                                        const float* __restrict__ bias1,
                                                        bf16_t* __restrict__ C,
                                                        int M, int N, int K) {
    constexpr int BM = 128, BN = 128, BK = 32;
    __shared__ bf16_t sA[BM * BK];   // 8 KB
    __shared__ bf16_t sW[BN * BK];   // 8 KB

    const int tid  = threadIdx.x;
    const int lane = tid & 31;
    const int wid  = tid >> 5;         // 0..7
    const int wm   = wid & 3;          // M quadrant (32 rows)
    const int wn   = wid >> 2;         // N half (64 cols)
    const int m0   = blockIdx.y * BM;
    const int n0   = blockIdx.x * BN;
    const int nk   = K / BK;

    v8f acc[2][4];
    #pragma unroll
    for (int mi = 0; mi < 2; ++mi)
        #pragma unroll
        for (int ni = 0; ni < 4; ++ni) acc[mi][ni] = {};

    // each thread stages 2 x 16B of A and 2 x 16B of W per K-tile
    u32x4 pa[2], pw[2];
    #pragma unroll
    for (int j = 0; j < 2; ++j) {
        int l = tid + j * 256;                      // 0..511 linear 16B index
        int row = l >> 2, cu = l & 3;
        pa[j] = *(gu32x4*)(A + (long long)(m0 + row) * K + cu * 8);
        pw[j] = *(gu32x4*)(W + (long long)(n0 + row) * K + cu * 8);
    }

    for (int kt = 0; kt < nk; ++kt) {
        // commit staged registers to LDS
        #pragma unroll
        for (int j = 0; j < 2; ++j) {
            int l = tid + j * 256;
            int row = l >> 2, cu = l & 3;
            *(u32x4*)(sA + row * BK + cu * 8) = pa[j];
            *(u32x4*)(sW + row * BK + cu * 8) = pw[j];
        }
        __syncthreads();

        // prefetch next K-tile from global while WMMA consumes LDS
        if (kt + 1 < nk) {
            int k1 = (kt + 1) * BK;
            #pragma unroll
            for (int j = 0; j < 2; ++j) {
                int l = tid + j * 256;
                int row = l >> 2, cu = l & 3;
                pa[j] = *(gu32x4*)(A + (long long)(m0 + row) * K + k1 + cu * 8);
                pw[j] = *(gu32x4*)(W + (long long)(n0 + row) * K + k1 + cu * 8);
            }
        }

        v16bf aF[2], bF[4];
        #pragma unroll
        for (int mi = 0; mi < 2; ++mi)
            aF[mi] = load_a_frag(sA + (wm * 32 + mi * 16) * BK, BK, lane);
        #pragma unroll
        for (int ni = 0; ni < 4; ++ni)
            bF[ni] = load_b_frag_nk(sW + (wn * 64 + ni * 16) * BK, BK, lane);

        #pragma unroll
        for (int mi = 0; mi < 2; ++mi)
            #pragma unroll
            for (int ni = 0; ni < 4; ++ni)
                acc[mi][ni] = __builtin_amdgcn_wmma_f32_16x16x32_bf16(
                    false, aF[mi], false, bF[ni], (short)0, acc[mi][ni], false, false);
        __syncthreads();
    }

    // epilogue: bias (+optional second bias), optional leaky-relu, store bf16
    #pragma unroll
    for (int ni = 0; ni < 4; ++ni) {
        int col = n0 + wn * 64 + ni * 16 + (lane & 15);
        float bsum = bias0[col] + (bias1 ? bias1[col] : 0.0f);
        #pragma unroll
        for (int mi = 0; mi < 2; ++mi) {
            #pragma unroll
            for (int r = 0; r < 8; ++r) {
                int row = m0 + wm * 32 + mi * 16 + r + ((lane < 16) ? 0 : 8);
                float v = acc[mi][ni][r] + bsum;
                if (ACT == 1) v = (v > 0.0f) ? v : 0.01f * v;
                C[(long long)row * N + col] = (bf16_t)v;
            }
        }
    }
}

// -------------------------- LSTM recurrence ---------------------------------
// One 1024-thread workgroup per direction; h/c persistent in LDS for 512 steps.
// Per step: 32 waves x 4 gate N-tiles x 16 K-tiles of v_wmma_f32_16x16x32_bf16,
// gate exchange through LDS, then pointwise LSTM cell update.
// Wh is step-invariant and streamed from L2 each step. A 32-bit offset is
// laundered through volatile asm (so LICM cannot hoist the 2 MB working set
// out of the 512-step loop and spill it), while the loads themselves go
// through explicit addrspace(1) pointers so they lower to global_load_b128
// (not flat_load_b128, which would serialize against the LDS counter).
__global__ __launch_bounds__(1024) void lstm_recurrence_kernel(
        const bf16_t* __restrict__ XG,   // [2][BS][4H] pre-computed input gates (this layer)
        const bf16_t* __restrict__ Wh,   // [2][4H][H]  recurrent weights (this layer), [N,K]
        const int*    __restrict__ lengths,
        bf16_t*       __restrict__ Y) {  // [BS][2H]  masked hidden outputs
    const int dir  = blockIdx.x;
    const int tid  = threadIdx.x;
    const int lane = tid & 31;
    const int wid  = tid >> 5;           // 0..31
    const int nc0  = wid * 64;           // this wave's 64 gate columns

    extern __shared__ char smem[];
    bf16_t* h_lds = (bf16_t*)smem;                                   // [16][512]  16 KB
    float*  c_lds = (float*)(smem + kB * kH * sizeof(bf16_t));       // [16][512]  32 KB
    float*  g_lds = (float*)(smem + kB * kH * (sizeof(bf16_t) + sizeof(float))); // [16][2048] 128 KB

    const bf16_t* xg_d = XG + (long long)dir * kBS * kG;
    const bf16_t* wh_d = Wh + (long long)dir * kG * kH;

    // per-lane invariant base: row (nc0 + lane%16), K-chunk by lane half
    const bf16_t* wbase = wh_d + (long long)(nc0 + (lane & 15)) * kH
                        + ((lane < 16) ? 0 : 16);

    for (int i = tid; i < kB * kH; i += 1024) {
        h_lds[i] = (bf16_t)0.0f;
        c_lds[i] = 0.0f;
    }
    __syncthreads();

    for (int j = 0; j < kS; ++j) {
        const int t = dir ? (kS - 1 - j) : j;

        // ---- phase 1: gates += h @ Wh^T via WMMA (full EXEC, all waves) ----
        v8f acc[4];
        #pragma unroll
        for (int ni = 0; ni < 4; ++ni) acc[ni] = {};

        #pragma unroll 2
        for (int kt = 0; kt < kH / 32; ++kt) {
            v16bf aF = load_a_frag(h_lds + kt * 32, kH, lane);

            // launder the K offset: loads stay inside the step loop (re-read
            // from L2 each step) but keep global-pointer provenance
            unsigned int koff = kt * 32;
            asm volatile("" : "+v"(koff));
            const bf16_t* wkt = wbase + koff;

            if (kt + 1 < kH / 32)
                __builtin_prefetch(wbase + (kt + 1) * 32, 0, 3);  // global_prefetch

            // issue all 4 B-fragment loads as one clause, then 4 WMMAs
            union { u32x4 u[2]; v16bf v; } tb[4];
            #pragma unroll
            for (int ni = 0; ni < 4; ++ni) {
                // ni stride = 16 rows * kH halves = 16 KB byte offset (imm24 OK)
                const bf16_t* wrow = wkt + ni * 16 * kH;
                tb[ni].u[0] = *(gu32x4*)(wrow);
                tb[ni].u[1] = *(gu32x4*)(wrow + 8);
            }
            #pragma unroll
            for (int ni = 0; ni < 4; ++ni)
                acc[ni] = __builtin_amdgcn_wmma_f32_16x16x32_bf16(
                    false, aF, false, tb[ni].v, (short)0, acc[ni], false, false);
        }
        #pragma unroll
        for (int ni = 0; ni < 4; ++ni) {
            int col = nc0 + ni * 16 + (lane & 15);
            #pragma unroll
            for (int r = 0; r < 8; ++r) {
                int m = r + ((lane < 16) ? 0 : 8);          // batch row
                g_lds[m * kG + col] = acc[ni][r];
            }
        }
        __syncthreads();

        // ---- phase 2: pointwise cell update (8 units per thread) ----------
        for (int e = tid; e < kB * kH; e += 1024) {
            int b = e >> 9;          // batch
            int u = e & (kH - 1);    // hidden unit
            const bf16_t* xr = xg_d + (long long)(b * kS + t) * kG;
            float gi = sigmoidf_(g_lds[b * kG + u]            + (float)xr[u]);
            float gf = sigmoidf_(g_lds[b * kG + kH + u]       + (float)xr[kH + u]);
            float gg = tanhf    (g_lds[b * kG + 2 * kH + u]   + (float)xr[2 * kH + u]);
            float go = sigmoidf_(g_lds[b * kG + 3 * kH + u]   + (float)xr[3 * kH + u]);
            float cold = c_lds[b * kH + u];
            float cnew = gf * cold + gi * gg;
            float hnew = go * tanhf(cnew);
            bool  msk  = t < lengths[b];
            c_lds[b * kH + u] = msk ? cnew : cold;
            float hold = (float)h_lds[b * kH + u];
            h_lds[b * kH + u] = (bf16_t)(msk ? hnew : hold);
            Y[(long long)(b * kS + t) * (2 * kH) + dir * kH + u] = (bf16_t)(msk ? hnew : 0.0f);
        }
        __syncthreads();
    }
}

// -------------------------- tiny C=5 head -----------------------------------
__global__ __launch_bounds__(256) void head_kernel(const bf16_t* __restrict__ A2,
                                                   const float* __restrict__ W3,
                                                   const float* __restrict__ b3,
                                                   float* __restrict__ out) {
    int i = blockIdx.x * 256 + threadIdx.x;      // over 8192*5
    if (i >= kBS * kC) return;
    int c = i % kC, row = i / kC;
    const bf16_t* a = A2 + (long long)row * kHID;
    const float*  w = W3 + (long long)c * kHID;
    float acc = b3[c];
    #pragma unroll 8
    for (int k = 0; k < kHID; ++k) acc += (float)a[k] * w[k];
    out[i] = acc;
}

// ---------------------------------------------------------------------------
extern "C" void kernel_launch(void* const* d_in, const int* in_sizes, int n_in,
                              void* d_out, int out_size, void* d_ws, size_t ws_size,
                              hipStream_t stream) {
    const int*   x       = (const int*)d_in[0];
    const int*   lengths = (const int*)d_in[1];
    const float* lang    = (const float*)d_in[2];
    const float* etab    = (const float*)d_in[3];
    const float* Wih     = (const float*)d_in[4];   // [2][2][2048][1024]
    const float* Whh     = (const float*)d_in[5];   // [2][2][2048][512]
    const float* bih     = (const float*)d_in[6];   // [2][2][2048]
    const float* bhh     = (const float*)d_in[7];
    const float* W1      = (const float*)d_in[8];   // [1024][1024]
    const float* b1      = (const float*)d_in[9];
    const float* W2      = (const float*)d_in[10];
    const float* b2      = (const float*)d_in[11];
    const float* W3      = (const float*)d_in[12];  // [5][1024]
    const float* b3      = (const float*)d_in[13];
    float* out = (float*)d_out;

    // ---- workspace layout (bytes) ----
    char* ws = (char*)d_ws;
    size_t off = 0;
    auto alloc = [&](size_t bytes) { char* p = ws + off; off += (bytes + 255) & ~(size_t)255; return p; };
    bf16_t* wi_bf = (bf16_t*)alloc((size_t)2 * 2 * kG * kDIN * 2);  // 16.8 MB
    bf16_t* wh_bf = (bf16_t*)alloc((size_t)2 * 2 * kG * kH * 2);    //  8.4 MB
    bf16_t* w1_bf = (bf16_t*)alloc((size_t)kHID * kHID * 2);        //  2.1 MB
    bf16_t* w2_bf = (bf16_t*)alloc((size_t)kHID * kHID * 2);        //  2.1 MB
    bf16_t* Xa    = (bf16_t*)alloc((size_t)kBS * kDIN * 2);         // 16.8 MB
    bf16_t* Xb    = (bf16_t*)alloc((size_t)kBS * kDIN * 2);         // 16.8 MB
    bf16_t* A1    = (bf16_t*)alloc((size_t)kBS * kHID * 2);         // 16.8 MB
    bf16_t* A2    = (bf16_t*)alloc((size_t)kBS * kHID * 2);         // 16.8 MB
    bf16_t* XG    = (bf16_t*)alloc((size_t)2 * kBS * kG * 2);       // 67.1 MB
    (void)ws_size; (void)in_sizes; (void)n_in; (void)out_size;

    auto cvt = [&](const float* s, bf16_t* d, int n) {
        cvt_f32_bf16_kernel<<<(n + 255) / 256, 256, 0, stream>>>(s, d, n);
    };
    cvt(Wih, wi_bf, 2 * 2 * kG * kDIN);
    cvt(Whh, wh_bf, 2 * 2 * kG * kH);
    cvt(W1,  w1_bf, kHID * kHID);
    cvt(W2,  w2_bf, kHID * kHID);

    // ---- embedding + concat -> Xa ----
    {
        long long n = (long long)kBS * kDIN;
        embed_concat_kernel<<<(unsigned)((n + 255) / 256), 256, 0, stream>>>(x, lang, etab, Xa);
    }

    const size_t rec_lds = (size_t)kB * kH * sizeof(bf16_t)
                         + (size_t)kB * kH * sizeof(float)
                         + (size_t)kB * kG * sizeof(float);   // 180224 B (<320 KB/WGP)

    bf16_t* Xcur = Xa;
    bf16_t* Xnxt = Xb;
    for (int l = 0; l < 2; ++l) {
        // gate pre-compute GEMMs, both directions: XG[d] = Xcur @ Wi^T + (bih+bhh)
        for (int d = 0; d < 2; ++d) {
            int ld = l * 2 + d;
            gemm_bf16_kernel<0><<<dim3(kG / 128, kBS / 128), 256, 0, stream>>>(
                Xcur, wi_bf + (size_t)ld * kG * kDIN,
                bih + (size_t)ld * kG, bhh + (size_t)ld * kG,
                XG + (size_t)d * kBS * kG, kBS, kG, kDIN);
        }
        // sequential recurrence, one workgroup per direction
        lstm_recurrence_kernel<<<2, 1024, rec_lds, stream>>>(
            XG, wh_bf + (size_t)l * 2 * kG * kH, lengths, Xnxt);
        bf16_t* t = Xcur; Xcur = Xnxt; Xnxt = t;
    }

    // ---- MLP: leaky_relu(Xcur@W1^T+b1) -> leaky_relu(@W2^T+b2) -> head ----
    gemm_bf16_kernel<1><<<dim3(kHID / 128, kBS / 128), 256, 0, stream>>>(
        Xcur, w1_bf, b1, nullptr, A1, kBS, kHID, kHID);
    gemm_bf16_kernel<1><<<dim3(kHID / 128, kBS / 128), 256, 0, stream>>>(
        A1, w2_bf, b2, nullptr, A2, kBS, kHID, kHID);
    head_kernel<<<(kBS * kC + 255) / 256, 256, 0, stream>>>(A2, W3, b3, out);
}
